// EdgeRefinement_20641612824907
// MI455X (gfx1250) — matrix operand order repaired and checked by reading, not compiled, not run on hardware
//
#include <hip/hip_runtime.h>

#define BB 2
#define HH 480
#define WW 640
#define TS 32
#define HL 4
#define LT 40  // TS + 2*HL

typedef __attribute__((ext_vector_type(16))) _Float16 v16h;
typedef __attribute__((ext_vector_type(8)))  _Float16 v8h;
typedef __attribute__((ext_vector_type(8)))  float    v8f;
typedef __attribute__((ext_vector_type(4)))  unsigned int v4u;
typedef __attribute__((ext_vector_type(8)))  int v8i;
typedef __attribute__((ext_vector_type(4)))  int v4i;

#if defined(__has_builtin)
#if __has_builtin(__builtin_amdgcn_tensor_load_to_lds) && __has_builtin(__builtin_amdgcn_s_wait_tensorcnt)
#define HAVE_TDM 1
#endif
#endif
#if defined(__has_include)
#if __has_include(<hip/amd_detail/amd_gfx1250_TDM.h>)
#define TDM_6ARG 1
#endif
#endif

struct WPtrs { const float* w[7]; };

// ---------------------------------------------------------------------------
// Swizzle conv weights into the WMMA 16-bit A-fragment layout.
//   lane<16 : M=lane,    VGPR v<4 holds K=2v,2v+1 ; v>=4 holds K=16+2(v-4),+1
//   lane>=16: M=lane-16, VGPR v<4 holds K=8+2v,+1 ; v>=4 holds K=24+2(v-4),+1
// Stored as [ (layer*2+blk)*9 + tap ][ lane ][ 16 halves ].
// ---------------------------------------------------------------------------
__global__ void k_swz(WPtrs wp, _Float16* __restrict__ swz) {
  int i = blockIdx.x * blockDim.x + threadIdx.x;
  if (i >= 7 * 2 * 9 * 32 * 8) return;
  int v    = i & 7;
  int lane = (i >> 3) & 31;
  int t    = (i >> 8) % 9;
  int blk  = (i / (256 * 9)) & 1;
  int l    = i / (256 * 9 * 2);
  int m  = lane & 15;
  int hi = lane >> 4;
  int k0 = (v < 4 ? 0 : 16) + (hi ? 8 : 0) + 2 * (v & 3);
  int oc = blk * 16 + m;
  int ICr = (l == 0) ? 4 : 32;
  int OCr = (l == 6) ? 8 : 32;
  const float* w = wp.w[l];
  float f0 = (oc < OCr && (k0    ) < ICr) ? w[(oc * ICr + k0    ) * 9 + t] : 0.f;
  float f1 = (oc < OCr && (k0 + 1) < ICr) ? w[(oc * ICr + k0 + 1) * 9 + t] : 0.f;
  size_t base = (((size_t)(l * 2 + blk) * 9 + t) * 32 + lane) * 16 + (size_t)v * 2;
  swz[base]     = (_Float16)f0;
  swz[base + 1] = (_Float16)f1;
}

// ---------------------------------------------------------------------------
// Build padded fp16 channel-last input: ch0..2 = image*(2/256), ch3 = edge,
// ch4..31 = 0.  Layout: act[((b*H + y)*W + x)*32 + ch]
// ---------------------------------------------------------------------------
__global__ void k_prep(const float* __restrict__ img, const float* __restrict__ edge,
                       _Float16* __restrict__ act) {
  int i = blockIdx.x * blockDim.x + threadIdx.x;
  if (i >= BB * HH * WW) return;
  int x = i % WW;
  int y = (i / WW) % HH;
  int b = i / (WW * HH);
  _Float16* p = act + (size_t)i * 32;
  size_t ip = ((size_t)(b * 3) * HH + y) * WW + x;
  p[0] = (_Float16)(img[ip] * (2.0f / 256.0f));
  p[1] = (_Float16)(img[ip + (size_t)HH * WW] * (2.0f / 256.0f));
  p[2] = (_Float16)(img[ip + 2 * (size_t)HH * WW] * (2.0f / 256.0f));
  p[3] = (_Float16)edge[i];
  for (int c = 4; c < 32; ++c) p[c] = (_Float16)0.f;
}

// ---------------------------------------------------------------------------
// B-fragment gather: 32 ic x 16 px tile, channel-last fp16 activations.
// Lanes 0-15 hold K=0-7,16-23; lanes 16-31 hold K=8-15,24-31 of pixels 0-15.
// ---------------------------------------------------------------------------
__device__ __forceinline__ v16h load_bfrag(const _Float16* __restrict__ in,
                                           int b, int sy, int sx, int ch0) {
  v16h bm;
  if ((unsigned)sy < (unsigned)HH && (unsigned)sx < (unsigned)WW) {
    const v8h* p = (const v8h*)(in + (((size_t)(b * HH + sy)) * WW + sx) * 32 + ch0);
    v8h lo = p[0], hh = p[2];
    for (int q = 0; q < 8; ++q) { bm[q] = lo[q]; bm[8 + q] = hh[q]; }
  } else {
    for (int q = 0; q < 16; ++q) bm[q] = (_Float16)0.f;
  }
  return bm;
}

// ---------------------------------------------------------------------------
// One conv layer as implicit GEMM via v_wmma_f32_16x16x32_f16.
// Block = 4 waves (128 threads); each wave owns 32 pixels of one row and both
// 16-oc blocks: 2 B fragments x 2 A fragments -> 4 independent WMMA chains,
// 36 WMMAs/wave.  Layer weights (18,432 B) are DMA'd into LDS by the Tensor
// Data Mover (tensor_load_to_lds + s_wait_tensorcnt) when available.
// final_mode: emit clip(conv + bias + edge, 0, 1) into planar (B,8,H,W) fp32.
// ---------------------------------------------------------------------------
__global__ __launch_bounds__(128) void k_conv(const _Float16* __restrict__ in,
                                              _Float16* __restrict__ out,
                                              const _Float16* __restrict__ swz,
                                              const float* __restrict__ bias, int nbias,
                                              int dil, int relu, int final_mode,
                                              const float* __restrict__ edge,
                                              float* __restrict__ wout) {
  __shared__ _Float16 wl[9216];   // 2 blk * 9 taps * 32 lanes * 16 halves
  __shared__ float    bl[32];
  int tid = threadIdx.x;

#ifdef HAVE_TDM
  if (tid < 32) {
    // Tensor DMA descriptor: 1 row of 4608 dwords (18,432 B), global -> LDS.
    v4u g0;
    v8i g1;
    v4i g2 = {0, 0, 0, 0}, g3 = {0, 0, 0, 0};
    unsigned long long ga = (unsigned long long)(uintptr_t)swz;
    unsigned int lds_off = (unsigned int)(uintptr_t)wl;   // LDS aperture: addr[31:0]
    g0[0] = 1u;                                  // count = 1 valid descriptor
    g0[1] = lds_off;                             // lds_addr
    g0[2] = (unsigned int)ga;                    // global_addr[31:0]
    g0[3] = (unsigned int)(ga >> 32) | (2u << 30); // global_addr[56:32] | type=2
    g1[0] = (int)(2u << 16);                     // workgroup_mask=0, data_size=4B
    g1[1] = (int)(4608u << 16);                  // tensor_dim0[15:0]
    g1[2] = (int)(1u << 16);                     // tensor_dim0[31:16]=0, tensor_dim1[15:0]=1
    g1[3] = (int)(4608u << 16);                  // tensor_dim1 hi=0, tile_dim0=4608
    g1[4] = 1;                                   // tile_dim1=1, tile_dim2=0
    g1[5] = 4608;                                // tensor_dim0_stride[31:0]
    g1[6] = 0;
    g1[7] = 0;
#ifdef TDM_6ARG
    v8i g4 = {0, 0, 0, 0, 0, 0, 0, 0};
    __builtin_amdgcn_tensor_load_to_lds(g0, g1, g2, g3, g4, 0);
#else
    __builtin_amdgcn_tensor_load_to_lds(g0, g1, g2, g3, 0);
#endif
    __builtin_amdgcn_s_wait_tensorcnt(0);
  }
#else
  {
    const uint32_t* s32 = (const uint32_t*)swz;
    uint32_t* d32 = (uint32_t*)wl;
    for (int i = tid; i < 4608; i += 128) d32[i] = s32[i];
  }
#endif
  if (tid < 32) bl[tid] = (tid < nbias) ? bias[tid] : 0.f;
  __syncthreads();

  int wave = tid >> 5, lane = tid & 31;
  int y  = blockIdx.y;
  int b  = blockIdx.z;
  int px0 = ((blockIdx.x * 4 + wave) << 5) + (lane & 15);  // pixel group 0
  int px1 = px0 + 16;                                      // pixel group 1
  int hi = lane >> 4;
  int ch0 = hi ? 8 : 0;

  v8f acc00 = {}, acc01 = {}, acc10 = {}, acc11 = {};
  for (int t = 0; t < 9; ++t) {
    int sy = y + (t / 3 - 1) * dil;
    int dx = (t % 3 - 1) * dil;
    v16h bm0 = load_bfrag(in, b, sy, px0 + dx, ch0);
    v16h bm1 = load_bfrag(in, b, sy, px1 + dx, ch0);
    v16h a0 = *(const v16h*)(wl + ((size_t)((0 * 9 + t) * 32 + lane)) * 16);
    v16h a1 = *(const v16h*)(wl + ((size_t)((1 * 9 + t) * 32 + lane)) * 16);
    acc00 = __builtin_amdgcn_wmma_f32_16x16x32_f16(false, a0, false, bm0, (short)0, acc00, false, false);
    acc10 = __builtin_amdgcn_wmma_f32_16x16x32_f16(false, a1, false, bm0, (short)0, acc10, false, false);
    acc01 = __builtin_amdgcn_wmma_f32_16x16x32_f16(false, a0, false, bm1, (short)0, acc01, false, false);
    acc11 = __builtin_amdgcn_wmma_f32_16x16x32_f16(false, a1, false, bm1, (short)0, acc11, false, false);
  }

  if (!final_mode) {
    // C/D layout: VGPR r -> oc = blkBase + r + hi*8 ; 8 consecutive channels/lane
    v8h o00, o10, o01, o11;
    for (int r = 0; r < 8; ++r) {
      float b0 = bl[hi * 8 + r], b1 = bl[16 + hi * 8 + r];
      float v00 = acc00[r] + b0, v10 = acc10[r] + b1;
      float v01 = acc01[r] + b0, v11 = acc11[r] + b1;
      if (relu) {
        v00 = fmaxf(v00, 0.f); v10 = fmaxf(v10, 0.f);
        v01 = fmaxf(v01, 0.f); v11 = fmaxf(v11, 0.f);
      }
      o00[r] = (_Float16)v00; o10[r] = (_Float16)v10;
      o01[r] = (_Float16)v01; o11[r] = (_Float16)v11;
    }
    _Float16* ob0 = out + (((size_t)(b * HH + y)) * WW + px0) * 32;
    _Float16* ob1 = out + (((size_t)(b * HH + y)) * WW + px1) * 32;
    *(v8h*)(ob0 + hi * 8)      = o00;
    *(v8h*)(ob0 + 16 + hi * 8) = o10;
    *(v8h*)(ob1 + hi * 8)      = o01;
    *(v8h*)(ob1 + 16 + hi * 8) = o11;
  } else {
    if (hi == 0) {  // lanes 0..15 hold oc 0..7 of block 0 — all we need
      float e0 = edge[((size_t)b * HH + y) * WW + px0];
      float e1 = edge[((size_t)b * HH + y) * WW + px1];
      for (int r = 0; r < 8; ++r) {
        float v0 = fminf(fmaxf(acc00[r] + bl[r] + e0, 0.f), 1.f);
        float v1 = fminf(fmaxf(acc01[r] + bl[r] + e1, 0.f), 1.f);
        size_t rowp = (((size_t)(b * 8 + r)) * HH + y) * WW;
        wout[rowp + px0] = v0;
        wout[rowp + px1] = v1;
      }
    }
  }
}

// ---------------------------------------------------------------------------
// One depth propagation iteration (4 directional passes) on a 32x32 tile with
// halo 4, ping-ponging in LDS.  OOB halo loads 0 data + 0 weight, which exactly
// matches the reference's zero-pad shift semantics.
// ---------------------------------------------------------------------------
__global__ __launch_bounds__(256) void k_prop_d(const float* __restrict__ din,
                                                const float* __restrict__ w8,
                                                float* __restrict__ dout) {
  __shared__ float ta[LT][LT], tb[LT][LT];
  __shared__ float wm[4][LT][LT];
  int x0 = blockIdx.x * TS - HL, y0 = blockIdx.y * TS - HL;
  int b = blockIdx.z;
  size_t plane = (size_t)HH * WW;
  const float* wb = w8 + (size_t)b * 8 * plane;

  for (int i = threadIdx.x; i < LT * LT; i += 256) {
    int r = i / LT, c = i - r * LT;
    int gy = y0 + r, gx = x0 + c;
    bool ok = (unsigned)gy < (unsigned)HH && (unsigned)gx < (unsigned)WW;
    size_t pix = (size_t)gy * WW + gx;
    ta[r][c]    = ok ? din[(size_t)b * plane + pix] : 0.f;
    wm[0][r][c] = ok ? wb[0 * plane + pix] : 0.f;
    wm[1][r][c] = ok ? wb[1 * plane + pix] : 0.f;
    wm[2][r][c] = ok ? wb[2 * plane + pix] : 0.f;
    wm[3][r][c] = ok ? wb[3 * plane + pix] : 0.f;
  }
  __syncthreads();
  for (int i = threadIdx.x; i < LT * LT; i += 256) {       // L->R (x-1)
    int r = i / LT, c = i - r * LT;
    float d = ta[r][c], w = wm[0][r][c];
    tb[r][c] = (c > 0) ? (ta[r][c - 1] * w + d * (1.f - w)) : d;
  }
  __syncthreads();
  for (int i = threadIdx.x; i < LT * LT; i += 256) {       // R->L (x+1)
    int r = i / LT, c = i - r * LT;
    float d = tb[r][c], w = wm[1][r][c];
    ta[r][c] = (c < LT - 1) ? (tb[r][c + 1] * w + d * (1.f - w)) : d;
  }
  __syncthreads();
  for (int i = threadIdx.x; i < LT * LT; i += 256) {       // U->D (y-1)
    int r = i / LT, c = i - r * LT;
    float d = ta[r][c], w = wm[2][r][c];
    tb[r][c] = (r > 0) ? (ta[r - 1][c] * w + d * (1.f - w)) : d;
  }
  __syncthreads();
  for (int i = threadIdx.x; i < LT * LT; i += 256) {       // D->U (y+1)
    int r = i / LT, c = i - r * LT;
    float d = tb[r][c], w = wm[3][r][c];
    ta[r][c] = (r < LT - 1) ? (tb[r + 1][c] * w + d * (1.f - w)) : d;
  }
  __syncthreads();
  for (int i = threadIdx.x; i < TS * TS; i += 256) {
    int r = i / TS + HL, c = i % TS + HL;
    dout[(size_t)b * plane + (size_t)(y0 + r) * WW + (x0 + c)] = ta[r][c];
  }
}

// ---------------------------------------------------------------------------
// One normals propagation iteration (3 channels) + fused normalization.
// ---------------------------------------------------------------------------
__global__ __launch_bounds__(256) void k_prop_n(const float* __restrict__ nin,
                                                const float* __restrict__ w8,
                                                float* __restrict__ nout) {
  __shared__ float na[3][LT][LT], nb[3][LT][LT];
  __shared__ float wm[4][LT][LT];   // total static LDS = 10*6400 = 64000 B
  int x0 = blockIdx.x * TS - HL, y0 = blockIdx.y * TS - HL;
  int b = blockIdx.z;
  size_t plane = (size_t)HH * WW;
  const float* wb = w8 + (size_t)b * 8 * plane;

  for (int i = threadIdx.x; i < LT * LT; i += 256) {
    int r = i / LT, c = i - r * LT;
    int gy = y0 + r, gx = x0 + c;
    bool ok = (unsigned)gy < (unsigned)HH && (unsigned)gx < (unsigned)WW;
    size_t pix = (size_t)gy * WW + gx;
    for (int ch = 0; ch < 3; ++ch)
      na[ch][r][c] = ok ? nin[((size_t)(b * 3 + ch)) * plane + pix] : 0.f;
    for (int j = 0; j < 4; ++j)
      wm[j][r][c] = ok ? wb[(size_t)(4 + j) * plane + pix] : 0.f;
  }
  __syncthreads();
  for (int i = threadIdx.x; i < LT * LT; i += 256) {       // L->R
    int r = i / LT, c = i - r * LT;
    float w = wm[0][r][c];
    for (int ch = 0; ch < 3; ++ch) {
      float d = na[ch][r][c];
      nb[ch][r][c] = (c > 0) ? (na[ch][r][c - 1] * w + d * (1.f - w)) : d;
    }
  }
  __syncthreads();
  for (int i = threadIdx.x; i < LT * LT; i += 256) {       // R->L
    int r = i / LT, c = i - r * LT;
    float w = wm[1][r][c];
    for (int ch = 0; ch < 3; ++ch) {
      float d = nb[ch][r][c];
      na[ch][r][c] = (c < LT - 1) ? (nb[ch][r][c + 1] * w + d * (1.f - w)) : d;
    }
  }
  __syncthreads();
  for (int i = threadIdx.x; i < LT * LT; i += 256) {       // U->D
    int r = i / LT, c = i - r * LT;
    float w = wm[2][r][c];
    for (int ch = 0; ch < 3; ++ch) {
      float d = na[ch][r][c];
      nb[ch][r][c] = (r > 0) ? (na[ch][r - 1][c] * w + d * (1.f - w)) : d;
    }
  }
  __syncthreads();
  for (int i = threadIdx.x; i < LT * LT; i += 256) {       // D->U
    int r = i / LT, c = i - r * LT;
    float w = wm[3][r][c];
    for (int ch = 0; ch < 3; ++ch) {
      float d = nb[ch][r][c];
      na[ch][r][c] = (r < LT - 1) ? (nb[ch][r + 1][c] * w + d * (1.f - w)) : d;
    }
  }
  __syncthreads();
  for (int i = threadIdx.x; i < TS * TS; i += 256) {
    int r = i / TS + HL, c = i % TS + HL;
    float nx = na[0][r][c], ny = na[1][r][c], nz = na[2][r][c];
    float s = fmaxf(sqrtf(nx * nx + ny * ny + nz * nz), 1e-12f);
    float inv = 1.f / s;
    size_t pix = (size_t)(y0 + r) * WW + (x0 + c);
    nout[((size_t)(b * 3 + 0)) * plane + pix] = nx * inv;
    nout[((size_t)(b * 3 + 1)) * plane + pix] = ny * inv;
    nout[((size_t)(b * 3 + 2)) * plane + pix] = nz * inv;
  }
}

// ---------------------------------------------------------------------------
extern "C" void kernel_launch(void* const* d_in, const int* in_sizes, int n_in,
                              void* d_out, int out_size, void* d_ws, size_t ws_size,
                              hipStream_t stream) {
  const float* image   = (const float*)d_in[0];
  const float* edge    = (const float*)d_in[1];
  const float* depth   = (const float*)d_in[2];
  const float* normals = (const float*)d_in[3];
  WPtrs wp;
  const float* bptr[7];
  for (int l = 0; l < 7; ++l) {
    wp.w[l] = (const float*)d_in[4 + 2 * l];
    bptr[l] = (const float*)d_in[5 + 2 * l];
  }

  char* ws = (char*)d_ws;
  const size_t actBytes = (size_t)BB * HH * WW * 32 * 2;   // 39,321,600 each
  _Float16* actA = (_Float16*)ws;
  _Float16* actB = (_Float16*)(ws + actBytes);
  _Float16* swz  = (_Float16*)(ws + 2 * actBytes);
  const size_t swzBytes = (size_t)7 * 2 * 9 * 512 * 2;     // 129,024
  float* wout = (float*)(ws + 2 * actBytes + ((swzBytes + 255) & ~(size_t)255));

  // Propagation ping-pong buffers reuse the actA region (free after last conv).
  float* dp0 = (float*)ws;
  float* dp1 = dp0 + (size_t)BB * HH * WW;
  float* np0 = dp1 + (size_t)BB * HH * WW;
  float* np1 = np0 + (size_t)BB * 3 * HH * WW;

  float* outDepth = (float*)d_out;
  float* outNorm  = (float*)d_out + (size_t)BB * HH * WW;

  // 1) weight swizzle + padded fp16 input
  k_swz<<<(7 * 2 * 9 * 32 * 8 + 255) / 256, 256, 0, stream>>>(wp, swz);
  k_prep<<<(BB * HH * WW + 255) / 256, 256, 0, stream>>>(image, edge, actA);

  // 2) 7 conv layers, ping-pong actA/actB; last layer emits clipped planar weights
  dim3 cgrid(WW / (32 * 4), HH, BB);   // (5, 480, 2), 128 threads = 4 waves
  const _Float16* cin = actA;
  _Float16* cout = actB;
  for (int l = 0; l < 7; ++l) {
    int dil   = (l < 3) ? 2 : 1;
    int relu  = (l < 6) ? 1 : 0;
    int fin   = (l == 6) ? 1 : 0;
    int nbias = (l == 6) ? 8 : 32;
    k_conv<<<cgrid, 128, 0, stream>>>(cin, cout, swz + (size_t)l * 9216, bptr[l],
                                      nbias, dil, relu, fin, edge, wout);
    const _Float16* t = cin; cin = cout; cout = (_Float16*)t;
  }

  // 3) propagation: 4 depth iterations, 4 normals iterations (w/ normalize)
  dim3 pgrid(WW / TS, HH / TS, BB);    // (20, 15, 2)
  k_prop_d<<<pgrid, 256, 0, stream>>>(depth, wout, dp0);
  k_prop_d<<<pgrid, 256, 0, stream>>>(dp0,   wout, dp1);
  k_prop_d<<<pgrid, 256, 0, stream>>>(dp1,   wout, dp0);
  k_prop_d<<<pgrid, 256, 0, stream>>>(dp0,   wout, outDepth);

  k_prop_n<<<pgrid, 256, 0, stream>>>(normals, wout, np0);
  k_prop_n<<<pgrid, 256, 0, stream>>>(np0,     wout, np1);
  k_prop_n<<<pgrid, 256, 0, stream>>>(np1,     wout, np0);
  k_prop_n<<<pgrid, 256, 0, stream>>>(np0,     wout, outNorm);
}
